// UiMatrix_9552007266412
// MI455X (gfx1250) — compile-verified, matrix operand order checked by reading, not run on hardware
//
// CDNA5 / gfx1250 implementation of the opinion-GNN forward pass.
// All dense 128-wide GEMMs run on v_wmma_f32_16x16x32_f16 (f16 in, f32 acc),
// with gathered rows staged in LDS and weight B-fragments held in registers.

#include <hip/hip_runtime.h>
#include <math.h>

#define A_ATTR 4
#define NUSER  16000
#define NITEM  10000
#define NOPI   5000
#define LSEQ   12
#define DIMD   128
#define LD     136   // f16 LDS row stride (halves): 16B-aligned frags, staggered banks

typedef __attribute__((ext_vector_type(16))) _Float16 v16h;
typedef __attribute__((ext_vector_type(8)))  _Float16 v8h;
typedef __attribute__((ext_vector_type(8)))  float    v8f;

__device__ __forceinline__ v8f wmma_f16(v16h a, v16h b, v8f c) {
  // (neg_a, A, neg_b, B, c_mod, C, reuse_a, reuse_b)
  return __builtin_amdgcn_wmma_f32_16x16x32_f16(false, a, false, b, (short)0, c,
                                                false, false);
}

// A-fragment (16x32 f16) from a row-major f16 LDS array with stride LD.
// Lane layout per ISA: lanes 0-15 hold K {0..7,16..23}, lanes 16-31 hold
// K {8..15,24..31} of row (lane&15).
__device__ __forceinline__ v16h frag_lds(const _Float16* base, int row0, int kc) {
  const int lane = threadIdx.x & 31;
  const _Float16* p = base + (row0 + (lane & 15)) * LD + kc + ((lane >> 4) << 3);
  v8h lo = *(const v8h*)p;
  v8h hi = *(const v8h*)(p + 16);
  return __builtin_shufflevector(lo, hi, 0, 1, 2, 3, 4, 5, 6, 7,
                                 8, 9, 10, 11, 12, 13, 14, 15);
}

// B-fragment for y = x @ W^T : B[k,n] = W[n,k]; lane n holds row n of W with
// the same K packing as the A fragment. Loaded from global f32, converted f16.
__device__ __forceinline__ v16h wfrag_g(const float* __restrict__ W, int n0, int kc) {
  const int lane = threadIdx.x & 31;
  const float* p = W + (size_t)(n0 + (lane & 15)) * DIMD + kc + ((lane >> 4) << 3);
  v16h f;
#pragma unroll
  for (int j = 0; j < 8; ++j) f[j] = (_Float16)p[j];
#pragma unroll
  for (int j = 0; j < 8; ++j) f[8 + j] = (_Float16)p[16 + j];
  return f;
}

// ---------------------------------------------------------------------------
// Kernel 1: opin_update for one side (user or item), one attribute per block.y.
// Block handles 8 nodes: 96 gathered rows -> q MLP (WMMA) -> k/v MLPs (WMMA)
// -> masked dots -> aggregation -> residual into workspace [A][N][128].
// ---------------------------------------------------------------------------
__global__ __launch_bounds__(256) void opin_update_kernel(
    const float* __restrict__ table, int ntab,       // neighbor chunk [A][ntab][D]
    const float* __restrict__ opin,                  // [NOPI][D]
    const float* __restrict__ attr_emb,              // [A][D]
    const float* __restrict__ self, int nself,       // [A][nself][D]
    const int* __restrict__ seqs,                    // [A][nself][L]
    const int* __restrict__ oseqs,                   // [A][nself][L]
    const float* __restrict__ qW, const float* __restrict__ qb,
    const float* __restrict__ kW, const float* __restrict__ kb,
    const float* __restrict__ vW, const float* __restrict__ vb,
    int wbase,
    float* __restrict__ out1)                        // [A][nself][D]
{
  extern __shared__ char smem[];
  float*    sQ   = (float*)smem;                 // 16*128 f32
  float*    sW   = sQ + 16 * DIMD;               // 96 f32
  int*      sSeq = (int*)(sW + 96);              // 96 int
  _Float16* sXq  = (_Float16*)(sSeq + 96);       // 16*LD
  _Float16* sXk  = sXq + 16 * LD;                // 96*LD
  _Float16* sXv  = sXk + 96 * LD;                // 96*LD
  _Float16* sKl  = sXv + 96 * LD;                // 96*LD
  _Float16* sVl  = sKl + 96 * LD;                // 96*LD

  const int a     = blockIdx.y;
  const int node0 = blockIdx.x * 8;
  const int tid   = threadIdx.x;
  const int widx  = wbase + a;

  const float* qWa = qW + (size_t)widx * DIMD * DIMD;
  const float* kWa = kW + (size_t)widx * DIMD * DIMD;
  const float* vWa = vW + (size_t)widx * DIMD * DIMD;
  const float* qba = qb + (size_t)widx * DIMD;
  const float* kba = kb + (size_t)widx * DIMD;
  const float* vba = vb + (size_t)widx * DIMD;
  const float* tab_a  = table + (size_t)a * ntab * DIMD;
  const float* self_a = self  + (size_t)a * nself * DIMD;
  const int*   seq_a  = seqs  + (size_t)a * nself * LSEQ;
  const int*   oseq_a = oseqs + (size_t)a * nself * LSEQ;

  // ---- gather + f32->f16 staging ----
  const int rt = tid >> 4;          // row-within-pass 0..15
  const int ct = (tid & 15) * 8;    // 8-float column slice
  float at[8];
#pragma unroll
  for (int j = 0; j < 8; ++j) at[j] = attr_emb[a * DIMD + ct + j];

#pragma unroll
  for (int pass = 0; pass < 6; ++pass) {
    const int r = pass * 16 + rt;   // 0..95 = node*LSEQ + l
    const int n = r / LSEQ;
    const int l = r - n * LSEQ;
    const int node = node0 + n;
    const int idx  = seq_a[(size_t)node * LSEQ + l];
    const int oidx = oseq_a[(size_t)node * LSEQ + l];
    if (ct == 0) sSeq[r] = idx;
    float nb[8] = {0.f, 0.f, 0.f, 0.f, 0.f, 0.f, 0.f, 0.f};
    float op[8] = {0.f, 0.f, 0.f, 0.f, 0.f, 0.f, 0.f, 0.f};
    if (idx > 0) {  // padded table row 0 == zeros
      const float4* p = (const float4*)(tab_a + (size_t)(idx - 1) * DIMD + ct);
      float4 f0 = p[0], f1 = p[1];
      nb[0] = f0.x; nb[1] = f0.y; nb[2] = f0.z; nb[3] = f0.w;
      nb[4] = f1.x; nb[5] = f1.y; nb[6] = f1.z; nb[7] = f1.w;
    }
    if (oidx > 0) {
      const float4* p = (const float4*)(opin + (size_t)(oidx - 1) * DIMD + ct);
      float4 f0 = p[0], f1 = p[1];
      op[0] = f0.x; op[1] = f0.y; op[2] = f0.z; op[3] = f0.w;
      op[4] = f1.x; op[5] = f1.y; op[6] = f1.z; op[7] = f1.w;
    }
    _Float16* xk = sXk + r * LD + ct;
    _Float16* xv = sXv + r * LD + ct;
#pragma unroll
    for (int j = 0; j < 8; ++j) {
      xk[j] = (_Float16)(at[j] + nb[j]);
      xv[j] = (_Float16)(nb[j] + op[j]);
    }
  }
  {  // q input: rows 0..7 = self embeddings, 8..15 zero pad
    float sv[8] = {0.f, 0.f, 0.f, 0.f, 0.f, 0.f, 0.f, 0.f};
    if (rt < 8) {
      const float4* p = (const float4*)(self_a + (size_t)(node0 + rt) * DIMD + ct);
      float4 f0 = p[0], f1 = p[1];
      sv[0] = f0.x; sv[1] = f0.y; sv[2] = f0.z; sv[3] = f0.w;
      sv[4] = f1.x; sv[5] = f1.y; sv[6] = f1.z; sv[7] = f1.w;
    }
    _Float16* xq = sXq + rt * LD + ct;
#pragma unroll
    for (int j = 0; j < 8; ++j) xq[j] = (_Float16)sv[j];
  }
  __syncthreads();

  // ---- WMMA GEMMs: each wave owns one 16-column output strip ----
  const int wave = tid >> 5, lane = tid & 31;
  const int n0   = wave * 16;
  const int col  = lane & 15;
  const int rsel = (lane >> 4) << 3;

  v16h fbq[4], fbk[4], fbv[4];
#pragma unroll
  for (int c = 0; c < 4; ++c) {
    fbq[c] = wfrag_g(qWa, n0, c * 32);
    fbk[c] = wfrag_g(kWa, n0, c * 32);
    fbv[c] = wfrag_g(vWa, n0, c * 32);
  }

  {  // q = tanh(self @ qW^T + qb), store f32
    v8f acc = {0.f, 0.f, 0.f, 0.f, 0.f, 0.f, 0.f, 0.f};
#pragma unroll
    for (int c = 0; c < 4; ++c) acc = wmma_f16(frag_lds(sXq, 0, c * 32), fbq[c], acc);
    const float b = qba[n0 + col];
#pragma unroll
    for (int j = 0; j < 8; ++j)
      sQ[(rsel + j) * DIMD + n0 + col] = tanhf(acc[j] + b);
  }
  {  // k = tanh(Xk @ kW^T + kb), store f16
    const float b = kba[n0 + col];
#pragma unroll
    for (int mt = 0; mt < 6; ++mt) {
      v8f acc = {0.f, 0.f, 0.f, 0.f, 0.f, 0.f, 0.f, 0.f};
#pragma unroll
      for (int c = 0; c < 4; ++c) acc = wmma_f16(frag_lds(sXk, mt * 16, c * 32), fbk[c], acc);
#pragma unroll
      for (int j = 0; j < 8; ++j)
        sKl[(mt * 16 + rsel + j) * LD + n0 + col] = (_Float16)tanhf(acc[j] + b);
    }
  }
  {  // v = tanh(Xv @ vW^T + vb), store f16
    const float b = vba[n0 + col];
#pragma unroll
    for (int mt = 0; mt < 6; ++mt) {
      v8f acc = {0.f, 0.f, 0.f, 0.f, 0.f, 0.f, 0.f, 0.f};
#pragma unroll
      for (int c = 0; c < 4; ++c) acc = wmma_f16(frag_lds(sXv, mt * 16, c * 32), fbv[c], acc);
#pragma unroll
      for (int j = 0; j < 8; ++j)
        sVl[(mt * 16 + rsel + j) * LD + n0 + col] = (_Float16)tanhf(acc[j] + b);
    }
  }
  __syncthreads();

  // ---- w[n,l] = (q . k) * (seq > 0) ----
  if (tid < 96) {
    const int n = tid / LSEQ;
    const float* q = sQ + n * DIMD;
    const _Float16* k = sKl + tid * LD;
    float acc = 0.f;
#pragma unroll
    for (int d = 0; d < DIMD; ++d) acc += q[d] * (float)k[d];
    sW[tid] = (sSeq[tid] > 0) ? acc : 0.f;
  }
  __syncthreads();

  // ---- out[n,:] = sum_l w[n,l] * v[n,l,:]; residual add into workspace ----
  {
    const int n  = tid >> 5;
    const int c0 = (tid & 31) * 4;
    float a0 = 0.f, a1 = 0.f, a2 = 0.f, a3 = 0.f;
#pragma unroll
    for (int l = 0; l < LSEQ; ++l) {
      const float wv = sW[n * LSEQ + l];
      const _Float16* v = sVl + (n * LSEQ + l) * LD + c0;
      a0 += wv * (float)v[0];
      a1 += wv * (float)v[1];
      a2 += wv * (float)v[2];
      a3 += wv * (float)v[3];
    }
    const int node = node0 + n;
    const float* e = self_a + (size_t)node * DIMD + c0;
    float* o = out1 + ((size_t)a * nself + node) * DIMD + c0;
    o[0] = e[0] + a0;
    o[1] = e[1] + a1;
    o[2] = e[2] + a2;
    o[3] = e[3] + a3;
  }
}

// ---------------------------------------------------------------------------
// Kernel 2: multi-head self-attention across A=4 attributes. 16 nodes/block
// (64 rows). Projections + concat-weight via WMMA; 4x4 masked softmax in VALU.
// ---------------------------------------------------------------------------
__global__ __launch_bounds__(256) void msa_kernel(
    const float* __restrict__ x1, int nnode,          // [A][nnode][D] (post-update)
    const int* __restrict__ seqs,                     // [A][nnode][L] (for mask)
    const float* __restrict__ Wq, const float* __restrict__ bq,
    const float* __restrict__ Wk, const float* __restrict__ bk,
    const float* __restrict__ Wv, const float* __restrict__ bv,
    const float* __restrict__ Wc,
    float* __restrict__ out)                          // [nnode][A][D]
{
  extern __shared__ char smem[];
  float*    sMask = (float*)smem;                 // 64 f32
  _Float16* sX = (_Float16*)(sMask + 64);         // 64*LD
  _Float16* sQ = sX + 64 * LD;
  _Float16* sK = sQ + 64 * LD;
  _Float16* sV = sK + 64 * LD;
  _Float16* sO = sX;                              // reuse X after projections

  const int node0 = blockIdx.x * 16;
  const int tid   = threadIdx.x;

  // ---- gather x (transpose [A][N][D] -> rows node*4+a) + mask ----
  const int rt = tid >> 4, ct = (tid & 15) * 8;
#pragma unroll
  for (int pass = 0; pass < 4; ++pass) {
    const int r = pass * 16 + rt;         // 0..63
    const int node = node0 + (r >> 2);
    const int a = r & 3;
    const float4* p = (const float4*)(x1 + ((size_t)a * nnode + node) * DIMD + ct);
    float4 f0 = p[0], f1 = p[1];
    _Float16* d = sX + r * LD + ct;
    d[0] = (_Float16)f0.x; d[1] = (_Float16)f0.y;
    d[2] = (_Float16)f0.z; d[3] = (_Float16)f0.w;
    d[4] = (_Float16)f1.x; d[5] = (_Float16)f1.y;
    d[6] = (_Float16)f1.z; d[7] = (_Float16)f1.w;
  }
  if (tid < 64) {
    const int node = node0 + (tid >> 2), a = tid & 3;
    const int* s = seqs + ((size_t)a * nnode + node) * LSEQ;
    int any = 0;
#pragma unroll
    for (int l = 0; l < LSEQ; ++l) any |= (s[l] > 0);
    sMask[tid] = any ? 1.f : 0.f;
  }
  __syncthreads();

  const int wave = tid >> 5, lane = tid & 31;
  const int n0   = wave * 16;
  const int col  = lane & 15;
  const int rsel = (lane >> 4) << 3;

  // ---- Q/K/V projections (no activation) ----
  v16h fq[4], fk[4], fv[4];
#pragma unroll
  for (int c = 0; c < 4; ++c) {
    fq[c] = wfrag_g(Wq, n0, c * 32);
    fk[c] = wfrag_g(Wk, n0, c * 32);
    fv[c] = wfrag_g(Wv, n0, c * 32);
  }
  const float bqv = bq[n0 + col], bkv = bk[n0 + col], bvv = bv[n0 + col];
#pragma unroll
  for (int mt = 0; mt < 4; ++mt) {
    v8f aq = {0.f, 0.f, 0.f, 0.f, 0.f, 0.f, 0.f, 0.f};
    v8f ak = {0.f, 0.f, 0.f, 0.f, 0.f, 0.f, 0.f, 0.f};
    v8f av = {0.f, 0.f, 0.f, 0.f, 0.f, 0.f, 0.f, 0.f};
#pragma unroll
    for (int c = 0; c < 4; ++c) {
      v16h xa = frag_lds(sX, mt * 16, c * 32);
      aq = wmma_f16(xa, fq[c], aq);
      ak = wmma_f16(xa, fk[c], ak);
      av = wmma_f16(xa, fv[c], av);
    }
#pragma unroll
    for (int j = 0; j < 8; ++j) {
      const int r = (mt * 16 + rsel + j) * LD + n0 + col;
      sQ[r] = (_Float16)(aq[j] + bqv);
      sK[r] = (_Float16)(ak[j] + bkv);
      sV[r] = (_Float16)(av[j] + bvv);
    }
  }
  __syncthreads();

  // ---- 4x4 attention per (node, head, a) with clamp-mask ----
  if (tid < 128) {
    const int node = tid >> 3;
    const int head = (tid >> 2) & 1;
    const int a    = tid & 3;
    const int rb = node * 4;
    const int d0 = head * 64;
    const float inva = 1.f - sMask[rb + a];
    float e[4], om[4];
    float mx = -3.0e38f;
#pragma unroll
    for (int b = 0; b < 4; ++b) {
      const _Float16* qp = sQ + (rb + a) * LD + d0;
      const _Float16* kp = sK + (rb + b) * LD + d0;
      float s = 0.f;
#pragma unroll
      for (int d = 0; d < 64; ++d) s += (float)qp[d] * (float)kp[d];
      const float invb = 1.f - sMask[rb + b];
      const float nm = fminf(1.f - inva * invb + (a == b ? 1.f : 0.f), 1.f);
      e[b]  = s * 0.125f - nm * 1e12f;   // 1/sqrt(hd=64) = 0.125
      om[b] = 1.f - nm;
      mx = fmaxf(mx, e[b]);
    }
    float w[4], se = 0.f;
#pragma unroll
    for (int b = 0; b < 4; ++b) { w[b] = expf(e[b] - mx); se += w[b]; }
    const float inv_se = 1.f / se;
#pragma unroll
    for (int b = 0; b < 4; ++b) w[b] = w[b] * inv_se * om[b];
#pragma unroll
    for (int d = 0; d < 64; ++d) {
      float o = 0.f;
#pragma unroll
      for (int b = 0; b < 4; ++b) o += w[b] * (float)sV[(rb + b) * LD + d0 + d];
      sO[(rb + a) * LD + d0 + d] = (_Float16)o;
    }
  }
  __syncthreads();

  // ---- output projection: out = O @ Wc^T (no bias), write [N][A][D] ----
  v16h fc[4];
#pragma unroll
  for (int c = 0; c < 4; ++c) fc[c] = wfrag_g(Wc, n0, c * 32);
#pragma unroll
  for (int mt = 0; mt < 4; ++mt) {
    v8f acc = {0.f, 0.f, 0.f, 0.f, 0.f, 0.f, 0.f, 0.f};
#pragma unroll
    for (int c = 0; c < 4; ++c) acc = wmma_f16(frag_lds(sO, mt * 16, c * 32), fc[c], acc);
#pragma unroll
    for (int j = 0; j < 8; ++j) {
      const int r = mt * 16 + rsel + j;   // local row = node*4+a
      out[((size_t)node0 * 4 + r) * DIMD + n0 + col] = acc[j];
    }
  }
}

// ---------------------------------------------------------------------------
// Host-side launcher. Input order = setup_inputs() dict order (params leaves
// in insertion order, then the four sequence arrays).
// ---------------------------------------------------------------------------
extern "C" void kernel_launch(void* const* d_in, const int* in_sizes, int n_in,
                              void* d_out, int out_size, void* d_ws, size_t ws_size,
                              hipStream_t stream) {
  const float* opin   = (const float*)d_in[0];   // [5000,128]
  const float* attr   = (const float*)d_in[1];   // [4,128]
  const float* uchunk = (const float*)d_in[2];   // [4,16000,128]
  const float* ichunk = (const float*)d_in[3];   // [4,10000,128]
  const float* qW = (const float*)d_in[4];       // [8,128,128]
  const float* qb = (const float*)d_in[5];       // [8,128]
  const float* kW = (const float*)d_in[6];
  const float* kb = (const float*)d_in[7];
  const float* vW = (const float*)d_in[8];
  const float* vb = (const float*)d_in[9];
  const float* uWq = (const float*)d_in[10];     // msa_u: Wq,bq,Wk,bk,Wv,bv,Wc
  const float* ubq = (const float*)d_in[11];
  const float* uWk = (const float*)d_in[12];
  const float* ubk = (const float*)d_in[13];
  const float* uWv = (const float*)d_in[14];
  const float* ubv = (const float*)d_in[15];
  const float* uWc = (const float*)d_in[16];
  const float* iWq = (const float*)d_in[17];     // msa_i
  const float* ibq = (const float*)d_in[18];
  const float* iWk = (const float*)d_in[19];
  const float* ibk = (const float*)d_in[20];
  const float* iWv = (const float*)d_in[21];
  const float* ibv = (const float*)d_in[22];
  const float* iWc = (const float*)d_in[23];
  const int* ui = (const int*)d_in[24];          // [4,16000,12]
  const int* uo = (const int*)d_in[25];
  const int* iu = (const int*)d_in[26];          // [4,10000,12]
  const int* io = (const int*)d_in[27];

  float* u1 = (float*)d_ws;                              // [A][NU][D]
  float* i1 = u1 + (size_t)A_ATTR * NUSER * DIMD;        // [A][NI][D]
  float* u_out = (float*)d_out;                          // [NU][A][D]
  float* i_out = u_out + (size_t)NUSER * A_ATTR * DIMD;  // [NI][A][D]

  const size_t updSmem =
      (16 * DIMD + 96) * sizeof(float) + 96 * sizeof(int) +
      (size_t)(16 * LD + 4 * 96 * LD) * sizeof(_Float16);   // 117,760 B
  const size_t msaSmem =
      64 * sizeof(float) + (size_t)4 * 64 * LD * sizeof(_Float16);  // 69,888 B

  // updates (UI_LAYERS == 1; both sides read the pre-update tables)
  opin_update_kernel<<<dim3(NUSER / 8, A_ATTR), 256, updSmem, stream>>>(
      ichunk, NITEM, opin, attr, uchunk, NUSER, ui, uo,
      qW, qb, kW, kb, vW, vb, /*wbase=*/0, u1);
  opin_update_kernel<<<dim3(NITEM / 8, A_ATTR), 256, updSmem, stream>>>(
      uchunk, NUSER, opin, attr, ichunk, NITEM, iu, io,
      qW, qb, kW, kb, vW, vb, /*wbase=*/A_ATTR, i1);

  // multi-head self-attention over attributes
  msa_kernel<<<NUSER / 16, 256, msaSmem, stream>>>(
      u1, NUSER, ui, uWq, ubq, uWk, ubk, uWv, ubv, uWc, u_out);
  msa_kernel<<<NITEM / 16, 256, msaSmem, stream>>>(
      i1, NITEM, iu, iWq, ibq, iWk, ibk, iWv, ibv, iWc, i_out);
}
